// VGG_68255620268319
// MI455X (gfx1250) — matrix operand order repaired and checked by reading, not compiled, not run on hardware
//
#include <hip/hip_runtime.h>
#include <stdint.h>
#include <stddef.h>

// ---------------------------------------------------------------------------
// Types for WMMA fragments (CDNA5 gfx1250, wave32)
// ---------------------------------------------------------------------------
typedef __attribute__((ext_vector_type(8)))  unsigned short u16x8;
typedef __attribute__((ext_vector_type(16))) __bf16         bf16x16;
typedef __attribute__((ext_vector_type(8)))  float          f32x8;
typedef __attribute__((ext_vector_type(4)))  unsigned int   u32x4;
typedef __attribute__((ext_vector_type(8)))  unsigned int   u32x8;

union Frag32 {            // 16 bf16 values = one A or B fragment per lane
    u16x8   h[2];
    bf16x16 v;
};

__device__ __forceinline__ unsigned short f2bf(float f) {
    unsigned int u = __float_as_uint(f);
    u += 0x7FFFu + ((u >> 16) & 1u);          // round-to-nearest-even
    return (unsigned short)(u >> 16);
}

// ---------------------------------------------------------------------------
// Input repack: fp32 NCHW [128,3,64,64] -> bf16 NHWC padded [128,64,64,32]
// ---------------------------------------------------------------------------
__global__ void pack_input(const float* __restrict__ x, unsigned short* __restrict__ act) {
    size_t idx = (size_t)blockIdx.x * 256 + threadIdx.x;
    const size_t total = (size_t)128 * 64 * 64 * 32;
    if (idx >= total) return;
    int c = (int)(idx & 31);
    size_t sp = idx >> 5;                 // n*4096 + iy*64 + ix
    int ix = (int)(sp & 63);
    int iy = (int)((sp >> 6) & 63);
    int n  = (int)(sp >> 12);
    float v = (c < 3) ? x[(((size_t)n * 3 + c) * 64 + iy) * 64 + ix] : 0.0f;
    act[idx] = f2bf(v);
}

// ---------------------------------------------------------------------------
// Weight repack: fp32 [Cout,Cin,3,3] -> bf16 Wp[kc][tap][half][Cout][16].
// For a fixed kc, the 18 (tap,half) rows are contiguous with constant row
// stride Cout*16 elements -> a single 2D TDM tile per (workgroup, kc).
// Per-lane B layout: lane = column n, lanes 0-15 hold K 0..15, lanes 16-31
// hold K 16..31, 16 contiguous K values per lane.
// ---------------------------------------------------------------------------
__global__ void pack_weight(const float* __restrict__ w, unsigned short* __restrict__ wp,
                            int Cout, int Cin, int KC) {
    int idx = blockIdx.x * 256 + threadIdx.x;
    int total = 9 * KC * 2 * Cout * 16;
    if (idx >= total) return;
    int j    = idx & 15;
    int t1   = idx >> 4;
    int n    = t1 % Cout;
    int t2   = t1 / Cout;
    int half = t2 & 1;
    int t3   = t2 >> 1;          // kc*9 + tap
    int tap  = t3 % 9;
    int kc   = t3 / 9;
    int c  = kc * 32 + half * 16 + j;
    int ky = tap / 3;
    int kx = tap - ky * 3;
    float v = (c < Cin) ? w[(((size_t)n * Cin + c) * 3 + ky) * 3 + kx] : 0.0f;
    wp[idx] = f2bf(v);
}

// ---------------------------------------------------------------------------
// Fold conv bias + eval-mode BN into per-channel scale/bias
// ---------------------------------------------------------------------------
__global__ void fold_bn(const float* __restrict__ cb, const float* __restrict__ g,
                        const float* __restrict__ bb, const float* __restrict__ m,
                        const float* __restrict__ v, float* __restrict__ scale,
                        float* __restrict__ bias, int C) {
    int c = blockIdx.x * 256 + threadIdx.x;
    if (c >= C) return;
    float s = g[c] * rsqrtf(v[c] + 1e-5f);
    scale[c] = s;
    bias[c]  = (cb[c] - m[c]) * s + bb[c];
}

// ---------------------------------------------------------------------------
// Fused conv3x3 (SAME) + BN + ReLU (+ optional 2x2 maxpool) via WMMA bf16.
//   Block: 128 threads (4 waves). Tile: 8x8 spatial x 64 Cout.
//   Per kc chunk: TDM stages the 18x2KB weight slice into LDS, threads stage
//   the 10x10x32 input halo tile into LDS, then 9 taps x 4 strips of WMMA.
//   Pooling is done in-register: horizontal max across accumulator pairs,
//   vertical max via ds_swizzle SWAPX16 (lane L <-> L^16 holds row dy+1).
// ---------------------------------------------------------------------------
template <int CIN_PAD, int COUT, int H, int W, bool POOL, bool LAST>
__global__ __launch_bounds__(128) void conv_bn_relu(
    const unsigned short* __restrict__ act,
    const unsigned short* __restrict__ wp,
    const float* __restrict__ scale,
    const float* __restrict__ bias,
    unsigned short* __restrict__ out_bf,
    float* __restrict__ out_f32) {
    constexpr int KC = CIN_PAD / 32;
    __shared__ unsigned short sW[9 * 2 * 64 * 16];   // 36864 B: weights for one kc
    __shared__ unsigned short sIn[10 * 10 * 32];     //  6400 B: input halo tile

    const int tid  = threadIdx.x;
    const int lane = tid & 31;
    const int wv   = tid >> 5;           // 0..3
    const int half = lane >> 4;          // 0/1
    const int l15  = lane & 15;

    constexpr int BXC = W / 8;
    constexpr int BYC = H / 8;
    const int b    = blockIdx.x;
    const int nimg = b / (BXC * BYC);
    const int rem  = b - nimg * (BXC * BYC);
    const int by   = (rem / BXC) * 8;
    const int bx   = (rem % BXC) * 8;
    const int nb   = blockIdx.y * 64;

    // A-fragment row owned by this lane (A layout: M = lane&15 for both halves)
    const int sA  = wv * 16 + l15;
    const int dyA = sA >> 3;
    const int dxA = sA & 7;

    f32x8 acc[4];
#pragma unroll
    for (int t = 0; t < 4; ++t)
#pragma unroll
        for (int r = 0; r < 8; ++r) acc[t][r] = 0.0f;

    u16x8 zero8;
#pragma unroll
    for (int i = 0; i < 8; ++i) zero8[i] = 0;

    // LDS byte offset = low 32 bits of the flat shared address (ISA 10.2)
    const unsigned int ldsW_addr = (unsigned int)(uintptr_t)(void*)sW;
    const unsigned long long wgbase =
        (unsigned long long)(uintptr_t)wp + (unsigned long long)nb * 16ull * 2ull;

    for (int kc = 0; kc < KC; ++kc) {
        if (kc) __syncthreads();   // previous LDS consumers done before refill

        if (tid == 0) {
            // ---- Tensor DMA: 2D tile, 18 rows x 512 dwords, row stride COUT*8 dwords
            unsigned long long ga =
                wgbase + (unsigned long long)kc * 18ull * COUT * 16ull * 2ull;
            u32x4 g0;
            g0[0] = 1u;                                   // count=1, user descriptor
            g0[1] = ldsW_addr;                            // lds_addr
            g0[2] = (unsigned int)ga;                     // global_addr[31:0]
            g0[3] = (unsigned int)((ga >> 32) & 0x01FFFFFFu) | (2u << 30); // [56:32] | type=2
            u32x8 g1;
            g1[0] = 0x00020000u;            // workgroup_mask=0, data_size=2 (4B)
            g1[1] = 0u;                     // tensor_dim0 = 0x40000000 -> low16 here = 0
            g1[2] = 0x4000u;                // tensor_dim0 hi16 ; tensor_dim1 low16 = 0
            g1[3] = 0x10u | (512u << 16);   // tensor_dim1 hi16 (=1M); tile_dim0 = 512
            g1[4] = 18u;                    // tile_dim1 = 18, tile_dim2 = 0
            g1[5] = (unsigned int)(COUT * 8); // tensor_dim0_stride (dword units)
            g1[6] = 0u;
            g1[7] = 0u;
            u32x4 gz;
            gz[0] = 0u; gz[1] = 0u; gz[2] = 0u; gz[3] = 0u;
            asm volatile("tensor_load_to_lds %0, %1, %2, %3"
                         :: "s"(g0), "s"(g1), "s"(gz), "s"(gz)
                         : "memory");
        }

        // ---- Stage 10x10x32 bf16 input halo tile (zeros outside the image)
        for (int i = tid; i < 400; i += 128) {
            int cell = i >> 2, pc = i & 3;
            int cy = cell / 10, cx = cell - cy * 10;
            int iy = by + cy - 1, ix = bx + cx - 1;
            u16x8 d = zero8;
            if (((unsigned)iy < (unsigned)H) && ((unsigned)ix < (unsigned)W))
                d = *(const u16x8*)(act +
                        ((size_t)(nimg * H + iy) * W + ix) * CIN_PAD + kc * 32 + pc * 8);
            *(u16x8*)(sIn + cell * 32 + pc * 8) = d;
        }

        if (tid < 32) __builtin_amdgcn_s_wait_tensorcnt(0);  // wave0: TDM done
        __syncthreads();

        // ---- 9 taps x 4 output strips of WMMA, all operands from LDS
#pragma unroll
        for (int tap = 0; tap < 9; ++tap) {
            const int ky = tap / 3;
            const int kx = tap - ky * 3;
            const unsigned short* ab =
                sIn + ((dyA + ky) * 10 + (dxA + kx)) * 32 + half * 16;
            Frag32 a;
            a.h[0] = *(const u16x8*)(ab);
            a.h[1] = *(const u16x8*)(ab + 8);
            const unsigned short* wb = sW + ((tap * 2 + half) * 64 + l15) * 16;
#pragma unroll
            for (int t = 0; t < 4; ++t) {
                Frag32 bm;
                bm.h[0] = *(const u16x8*)(wb + t * 256);
                bm.h[1] = *(const u16x8*)(wb + t * 256 + 8);
                acc[t] = __builtin_amdgcn_wmma_f32_16x16x32_bf16(
                    false, a.v, false, bm.v, (short)0, acc[t], false, false);
            }
        }
    }

    // ---- Epilogue: scale/bias + ReLU (+ fused 2x2 maxpool)
    // C/D layout: lane<16 -> N=lane, M=r (dy=wv*2, dx=r);
    //             lane>=16 -> N=lane-16, M=8+r (dy=wv*2+1, dx=r)
#pragma unroll
    for (int t = 0; t < 4; ++t) {
        const int n = nb + t * 16 + l15;
        const float sc = scale[n];
        const float bi = bias[n];
        float v[8];
#pragma unroll
        for (int r = 0; r < 8; ++r) v[r] = fmaxf(acc[t][r] * sc + bi, 0.0f);

        if (LAST) {
#pragma unroll
            for (int r = 0; r < 8; ++r) {
                const int s = wv * 16 + r + half * 8;
                out_f32[(((size_t)nimg * COUT + n) * H + by + (s >> 3)) * W +
                        bx + (s & 7)] = v[r];
            }
        } else if (POOL) {
#pragma unroll
            for (int j = 0; j < 4; ++j) {
                float p = fmaxf(v[2 * j], v[2 * j + 1]);             // horizontal max
                float q = __int_as_float(__builtin_amdgcn_ds_swizzle(
                    __float_as_int(p), 0x401F));                     // SWAPX16: row dy^1
                p = fmaxf(p, q);                                     // vertical max
                if (half == 0) {
                    const int oy = (by >> 1) + wv;
                    const int ox = (bx >> 1) + j;
                    out_bf[((size_t)(nimg * (H / 2) + oy) * (W / 2) + ox) * COUT + n] =
                        f2bf(p);
                }
            }
        } else {
#pragma unroll
            for (int r = 0; r < 8; ++r) {
                const int s = wv * 16 + r + half * 8;
                out_bf[((size_t)(nimg * H + by + (s >> 3)) * W + bx + (s & 7)) * COUT +
                       n] = f2bf(v[r]);
            }
        }
    }
}

// ---------------------------------------------------------------------------
// Launcher
// ---------------------------------------------------------------------------
static inline int cdiv_ll(long long a, int b) { return (int)((a + b - 1) / b); }

extern "C" void kernel_launch(void* const* d_in, const int* in_sizes, int n_in,
                              void* d_out, int out_size, void* d_ws, size_t ws_size,
                              hipStream_t stream) {
    (void)in_sizes; (void)n_in; (void)out_size; (void)ws_size;

    const float* x = (const float*)d_in[0];
    const float *cw[6], *cb[6], *bg[6], *bbp[6], *bm[6], *bv[6];
    for (int i = 0; i < 6; ++i) {
        cw[i]  = (const float*)d_in[1 + i * 6 + 0];
        cb[i]  = (const float*)d_in[1 + i * 6 + 1];
        bg[i]  = (const float*)d_in[1 + i * 6 + 2];
        bbp[i] = (const float*)d_in[1 + i * 6 + 3];
        bm[i]  = (const float*)d_in[1 + i * 6 + 4];
        bv[i]  = (const float*)d_in[1 + i * 6 + 5];
    }

    static const int Cin[6]  = {3, 64, 128, 256, 256, 512};
    static const int Cpad[6] = {32, 64, 128, 256, 256, 512};
    static const int Cout[6] = {64, 128, 256, 256, 512, 512};

    char* ws = (char*)d_ws;
    unsigned short* bufA = (unsigned short*)ws;                       // <= 33.6 MB live
    unsigned short* bufB = (unsigned short*)(ws + (size_t)33554432);  // <= 16.8 MB live
    char* p = ws + (size_t)33554432 + (size_t)16777216;

    unsigned short* wpk[6];
    for (int i = 0; i < 6; ++i) {
        wpk[i] = (unsigned short*)p;
        p += (size_t)9 * Cpad[i] * Cout[i] * 2;
    }
    float *scl[6], *bia[6];
    for (int i = 0; i < 6; ++i) {
        scl[i] = (float*)p; p += (size_t)Cout[i] * 4;
        bia[i] = (float*)p; p += (size_t)Cout[i] * 4;
    }

    // Stage 0: repack input + weights, fold BN
    pack_input<<<cdiv_ll((long long)128 * 64 * 64 * 32, 256), 256, 0, stream>>>(x, bufA);
    for (int i = 0; i < 6; ++i) {
        int KC  = Cpad[i] / 32;
        int tot = 9 * Cpad[i] * Cout[i];
        pack_weight<<<cdiv_ll(tot, 256), 256, 0, stream>>>(cw[i], wpk[i], Cout[i], Cin[i], KC);
        fold_bn<<<cdiv_ll(Cout[i], 256), 256, 0, stream>>>(cb[i], bg[i], bbp[i], bm[i], bv[i],
                                                           scl[i], bia[i], Cout[i]);
    }

    // L1: [128,64,64,32] -> pool-fused -> [128,32,32,64]
    conv_bn_relu<32, 64, 64, 64, true, false>
        <<<dim3(128 * 8 * 8, 1), 128, 0, stream>>>(bufA, wpk[0], scl[0], bia[0], bufB, nullptr);
    // L2: -> pool-fused -> [128,16,16,128]
    conv_bn_relu<64, 128, 32, 32, true, false>
        <<<dim3(128 * 4 * 4, 2), 128, 0, stream>>>(bufB, wpk[1], scl[1], bia[1], bufA, nullptr);
    // L3: -> [128,16,16,256]
    conv_bn_relu<128, 256, 16, 16, false, false>
        <<<dim3(128 * 2 * 2, 4), 128, 0, stream>>>(bufA, wpk[2], scl[2], bia[2], bufB, nullptr);
    // L4: -> pool-fused -> [128,8,8,256]
    conv_bn_relu<256, 256, 16, 16, true, false>
        <<<dim3(128 * 2 * 2, 4), 128, 0, stream>>>(bufB, wpk[3], scl[3], bia[3], bufA, nullptr);
    // L5: -> [128,8,8,512]
    conv_bn_relu<256, 512, 8, 8, false, false>
        <<<dim3(128, 8), 128, 0, stream>>>(bufA, wpk[4], scl[4], bia[4], bufB, nullptr);
    // L6: -> fp32 NCHW [128,512,8,8] directly to d_out
    conv_bn_relu<512, 512, 8, 8, false, true>
        <<<dim3(128, 8), 128, 0, stream>>>(bufB, wpk[5], scl[5], bia[5], nullptr, (float*)d_out);
}
